// ExampleNet_39633958207815
// MI455X (gfx1250) — compile-verified, matrix operand order checked by reading, not run on hardware
//
#include <hip/hip_runtime.h>

// ---------------------------------------------------------------------------
// SubM sparse conv net on gfx1250 (MI455X, wave32, WMMA).
//   K1: swizzle w2 into CDNA5 WMMA B-fragment layout (f16)
//   K2: mask + BN + conv1(1->32) + ReLU -> padded f16 intermediate (VALU)
//   K3: conv2(32->64) as implicit GEMM via v_wmma_f32_16x16x32_f16,
//       halo tiles staged to LDS via GLOBAL_LOAD_ASYNC_TO_LDS_B128 (ASYNCcnt)
//       when available, bias+ReLU+mask epilogue, NCHW store
// ---------------------------------------------------------------------------

typedef __attribute__((ext_vector_type(16))) _Float16 v16h;
typedef __attribute__((ext_vector_type(8)))  _Float16 v8h;
typedef __attribute__((ext_vector_type(8)))  float    v8f;
typedef __attribute__((__vector_size__(16))) int      v4i_vs;   // int4, matches builtin pointee

#define BATCH 4
#define HH 512
#define WW 512
#define HP 514           // padded (1-px halo) for the f16 intermediate
#define WP 514
#define C1 32
#define C2 64

#if defined(__has_builtin)
#  if __has_builtin(__builtin_amdgcn_global_load_async_to_lds_b128)
#    define HAVE_ASYNC_COPY 1
#  endif
#  if __has_builtin(__builtin_amdgcn_s_wait_asynccnt)
#    define HAVE_WAIT_ASYNC 1
#  endif
#endif

// ---------------------------------------------------------------------------
// K1: pre-swizzle w2 (3,3,32,64 HWIO fp32) into per-lane WMMA B-fragments.
// Fragment order: frag[(tap*4 + jN)*32 + lane] = 16 contiguous halves.
// CDNA5 16-bit A/B layout: lane l holds row/col (l&15); half h maps to
//   K = (h<8 ? h : 8+h) + (l>=16 ? 8 : 0)
// ---------------------------------------------------------------------------
__global__ void k_prep_w2(const float* __restrict__ w2, _Float16* __restrict__ wsB) {
  int idx = blockIdx.x * blockDim.x + threadIdx.x;
  if (idx >= 9 * 4 * 32 * 16) return;
  int h = idx & 15;
  int l = (idx >> 4) & 31;
  int j = (idx >> 9) & 3;
  int t = idx >> 11;
  int k = ((h < 8) ? h : (8 + h)) + ((l >> 4) << 3);  // K in 0..31
  int n = 16 * j + (l & 15);                          // out channel
  wsB[idx] = (_Float16)w2[(t * C1 + k) * C2 + n];
}

// ---------------------------------------------------------------------------
// K2: mask + BN + conv1 + ReLU -> y1 (padded [B,514,514,32] f16, halo = 0)
// ---------------------------------------------------------------------------
__global__ void k_conv1(const float* __restrict__ x,
                        const float* __restrict__ bn_gamma,
                        const float* __restrict__ bn_beta,
                        const float* __restrict__ bn_mean,
                        const float* __restrict__ bn_var,
                        const float* __restrict__ w1,
                        const float* __restrict__ b1,
                        _Float16* __restrict__ y1) {
  int tid = blockIdx.x * blockDim.x + threadIdx.x;
  const int NPAD = BATCH * HP * WP;
  if (tid >= NPAD) return;
  int b   = tid / (HP * WP);
  int rem = tid % (HP * WP);
  int ph = rem / WP, pw = rem % WP;
  v8h* out = (v8h*)(y1 + (long)tid * C1);

  bool halo = (ph == 0) | (ph == HP - 1) | (pw == 0) | (pw == WP - 1);
  int h = ph - 1, w = pw - 1;
  float xc = 0.0f;
  if (!halo) xc = x[((long)(b * HH + h)) * WW + w];
  if (halo || xc == 0.0f) {                 // inactive site -> exact zeros
    v8h z = {};
    out[0] = z; out[1] = z; out[2] = z; out[3] = z;
    return;
  }

  float scale = bn_gamma[0] * rsqrtf(bn_var[0] + 1e-5f);
  float mean = bn_mean[0], beta = bn_beta[0];

  float a[9];
#pragma unroll
  for (int ky = 0; ky < 3; ky++) {
#pragma unroll
    for (int kx = 0; kx < 3; kx++) {
      int hn = h + ky - 1, wn = w + kx - 1;
      float xv = 0.0f;
      if (hn >= 0 && hn < HH && wn >= 0 && wn < WW)
        xv = x[((long)(b * HH + hn)) * WW + wn];
      // BN'd value only where the neighbor is active; else exact zero
      a[ky * 3 + kx] = (xv != 0.0f) ? ((xv - mean) * scale + beta) : 0.0f;
    }
  }

  float acc[C1];
#pragma unroll
  for (int c = 0; c < C1; c++) acc[c] = b1[c];
#pragma unroll
  for (int t = 0; t < 9; t++) {
    float av = a[t];
#pragma unroll
    for (int c = 0; c < C1; c++) acc[c] = fmaf(av, w1[t * C1 + c], acc[c]);
  }

  v8h o[4];
#pragma unroll
  for (int c = 0; c < C1; c++) o[c >> 3][c & 7] = (_Float16)fmaxf(acc[c], 0.0f);
  out[0] = o[0]; out[1] = o[1]; out[2] = o[2]; out[3] = o[3];
}

// ---------------------------------------------------------------------------
// K3: conv2 as implicit GEMM with WMMA.
// Block = 128 threads (4 waves). Block tile: one image row r, 64 consecutive
// pixels; each wave: M=16 pixels x N=64 channels, K = 9 taps * 32 ch.
// ---------------------------------------------------------------------------
__global__ void __launch_bounds__(128) k_conv2(const _Float16* __restrict__ y1,
                                               const _Float16* __restrict__ wsB,
                                               const float* __restrict__ x,
                                               const float* __restrict__ b2,
                                               float* __restrict__ out) {
  __shared__ __align__(16) _Float16 ldsY[3 * 66 * C1];   // 12672 B halo tile

  int bid = blockIdx.x;
  int b   = bid >> 12;          // 4096 blocks per image (512 rows * 8 tiles)
  int rem = bid & 4095;
  int r   = rem >> 3;           // output row
  int c0  = (rem & 7) << 6;     // output col base (multiple of 64)
  int tid = threadIdx.x;

  // ---- stage y1 rows r..r+2 (padded coords), cols c0..c0+65 into LDS ----
  // 3 rows * 66 cols * 32 ch = 6336 halves = 792 16-byte chunks.
  // Preferred path: async DMA straight into LDS (no VGPR round-trip),
  // tracked by ASYNCcnt; fallback: classic load+ds_store.
  for (int q = tid; q < 792; q += 128) {
    int row = q / 264;          // 264 chunks per staged row
    int cq  = q - row * 264;
    const _Float16* src =
        y1 + ((long)(b * HP + (r + row)) * WP + c0) * C1 + cq * 8;
    _Float16* dst = ldsY + (row * 264 + cq) * 8;
#ifdef HAVE_ASYNC_COPY
    __builtin_amdgcn_global_load_async_to_lds_b128(
        (__attribute__((address_space(1))) v4i_vs*)src,
        (__attribute__((address_space(3))) v4i_vs*)dst,
        /*imm offset*/ 0, /*cpol*/ 0);
#else
    *(v8h*)dst = *(const v8h*)src;
#endif
  }
#ifdef HAVE_ASYNC_COPY
#  ifdef HAVE_WAIT_ASYNC
  __builtin_amdgcn_s_wait_asynccnt(0);
#  else
  asm volatile("s_wait_asynccnt 0x0" ::: "memory");
#  endif
#endif
  __syncthreads();

  int lane = tid & 31;
  int wv   = tid >> 5;          // wave id 0..3 -> 16-pixel sub-tile
  int M    = lane & 15;
  int hi   = lane >> 4;         // half-wave selects K lo/hi groups
  int koff = hi << 3;

  v8f acc0 = {}, acc1 = {}, acc2 = {}, acc3 = {};

#pragma unroll
  for (int dy = 0; dy < 3; dy++) {
#pragma unroll
    for (int dx = 0; dx < 3; dx++) {
      int t   = dy * 3 + dx;
      int col = 16 * wv + M + dx;                 // LDS col of neighbor pixel
      const _Float16* ap = ldsY + (dy * 66 + col) * C1;
      union { v16h v; v8h h2[2]; } A;
      A.h2[0] = *(const v8h*)(ap + koff);         // K = koff .. koff+7
      A.h2[1] = *(const v8h*)(ap + 16 + koff);    // K = 16+koff .. 16+koff+7

      const v16h* bf = ((const v16h*)wsB) + (t * 4) * 32 + lane;
      v16h Bv;
      Bv = bf[0 * 32];
      acc0 = __builtin_amdgcn_wmma_f32_16x16x32_f16(false, A.v, false, Bv,
                                                    (short)0, acc0, false, false);
      Bv = bf[1 * 32];
      acc1 = __builtin_amdgcn_wmma_f32_16x16x32_f16(false, A.v, false, Bv,
                                                    (short)0, acc1, false, false);
      Bv = bf[2 * 32];
      acc2 = __builtin_amdgcn_wmma_f32_16x16x32_f16(false, A.v, false, Bv,
                                                    (short)0, acc2, false, false);
      Bv = bf[3 * 32];
      acc3 = __builtin_amdgcn_wmma_f32_16x16x32_f16(false, A.v, false, Bv,
                                                    (short)0, acc3, false, false);
    }
  }

  // ---- epilogue: bias + ReLU + submanifold mask, NCHW store ----
  // D-fragment layout: lane l -> N = l&15, VGPR i -> pixel M = i + 8*(l>=16)
  int n0    = lane & 15;
  int wbase = c0 + 16 * wv + 8 * hi;              // first pixel col this lane
  const float* xrow = x + ((long)(b * HH + r)) * WW + wbase;
  float m[8];
#pragma unroll
  for (int i = 0; i < 8; i++) m[i] = xrow[i];     // mask source: x != 0

  v8f accs[4] = {acc0, acc1, acc2, acc3};
#pragma unroll
  for (int j = 0; j < 4; j++) {
    int n = 16 * j + n0;
    float bias = b2[n];
    float* o = out + ((long)(b * C2 + n) * HH + r) * WW + wbase;
#pragma unroll
    for (int i = 0; i < 8; i++) {
      float v = fmaxf(accs[j][i] + bias, 0.0f);
      o[i] = (m[i] != 0.0f) ? v : 0.0f;
    }
  }
}

// ---------------------------------------------------------------------------
extern "C" void kernel_launch(void* const* d_in, const int* in_sizes, int n_in,
                              void* d_out, int out_size, void* d_ws, size_t ws_size,
                              hipStream_t stream) {
  const float* x        = (const float*)d_in[0];
  const float* bn_gamma = (const float*)d_in[1];
  const float* bn_beta  = (const float*)d_in[2];
  const float* bn_mean  = (const float*)d_in[3];
  const float* bn_var   = (const float*)d_in[4];
  const float* w1       = (const float*)d_in[5];
  const float* b1       = (const float*)d_in[6];
  const float* w2       = (const float*)d_in[7];
  const float* b2       = (const float*)d_in[8];

  // workspace: [0, 36864) = w2 B-fragments (f16); then padded y1 (f16, ~67.6MB)
  _Float16* wsB = (_Float16*)d_ws;
  _Float16* y1  = (_Float16*)((char*)d_ws + 36864);

  k_prep_w2<<<(9 * 4 * 32 * 16 + 255) / 256, 256, 0, stream>>>(w2, wsB);

  int npad = BATCH * HP * WP;
  k_conv1<<<(npad + 255) / 256, 256, 0, stream>>>(x, bn_gamma, bn_beta, bn_mean,
                                                  bn_var, w1, b1, y1);

  k_conv2<<<BATCH * HH * (WW / 64), 128, 0, stream>>>(y1, wsB, x, b2,
                                                      (float*)d_out);
}
